// SparseGroupedExperts_18451179504162
// MI455X (gfx1250) — matrix operand order repaired and checked by reading, not compiled, not run on hardware
//
#include <hip/hip_runtime.h>
#include <hip/hip_fp16.h>

typedef __attribute__((ext_vector_type(16))) _Float16 v16h;
typedef __attribute__((ext_vector_type(8)))  _Float16 v8h;
typedef __attribute__((ext_vector_type(4)))  _Float16 v4h;
typedef __attribute__((ext_vector_type(8)))  float    v8f;
typedef __attribute__((ext_vector_type(4)))  int      v4i;

#define N_EXPERTS 8
#define D_MODEL   1024
#define D_FF      2048
#define N_TOKENS  8192
#define TOP_K     2
#define SLOT_CAP  (N_TOKENS * TOP_K)   // 16384 routed slots total
#define LDK       48                   // padded LDS row stride in halves (96B, 16B-aligned)

// ---------------- gfx1250 async-to-LDS staging (guarded) ----------------
#if defined(__AMDGCN__) && defined(__has_builtin)
#if __has_builtin(__builtin_amdgcn_global_load_async_to_lds_b128) && \
    __has_builtin(__builtin_amdgcn_s_wait_asynccnt)
#define HAVE_ASYNC_LDS 1
#endif
#endif

#ifdef HAVE_ASYNC_LDS
typedef __attribute__((address_space(1))) v4i gas_v4i;  // global
typedef __attribute__((address_space(3))) v4i las_v4i;  // LDS
#endif

// Copy 16 bytes global -> LDS. Async path uses GLOBAL_LOAD_ASYNC_TO_LDS_B128
// (ASYNCcnt-tracked, no VGPR round trip); fallback is b128 load + ds_store_b128.
__device__ __forceinline__ void stage16(const _Float16* g, _Float16* l) {
#ifdef HAVE_ASYNC_LDS
  __builtin_amdgcn_global_load_async_to_lds_b128(
      (gas_v4i*)(const void*)g, (las_v4i*)(void*)l, 0, 0);
#else
  *(v8h*)l = *(const v8h*)g;
#endif
}
// Async loads complete in order per wave: waiting <=4 drains the previous
// 4-op chunk while the just-issued 4 ops stay in flight (double buffering).
__device__ __forceinline__ void stage_wait_le4() {
#ifdef HAVE_ASYNC_LDS
  __builtin_amdgcn_s_wait_asynccnt(4);
#endif
}
__device__ __forceinline__ void stage_wait_0() {
#ifdef HAVE_ASYNC_LDS
  __builtin_amdgcn_s_wait_asynccnt(0);
#endif
}

// ---------------- WMMA helpers ----------------

__device__ __forceinline__ v8f wmma_f16(v16h a, v16h b, v8f c) {
  // D = A(16x32 f16) * B(32x16 f16) + C(16x16 f32)
  return __builtin_amdgcn_wmma_f32_16x16x32_f16(false, a, false, b, (short)0, c, false, false);
}

// A fragment (16x32 f16). ISA: lane holds row M = lane%16; half = lane/16
// selects K in {8h..8h+7} U {16+8h..16+8h+7}.
__device__ __forceinline__ v16h load_a_frag(const _Float16* tile, int lane) {
  const int m = lane & 15, half = lane >> 4;
  const _Float16* p = tile + m * LDK + half * 8;
  v8h lo = *(const v8h*)(p);
  v8h hi = *(const v8h*)(p + 16);
  v16h r;
#pragma unroll
  for (int j = 0; j < 8; ++j) { r[j] = lo[j]; r[j + 8] = hi[j]; }
  return r;
}

// B fragment (32x16 f16) from an LDS tile stored TRANSPOSED as Bt[n][k].
// ISA: lane holds col N = lane%16; half = lane/16 selects K in {8h..8h+7} U {16+8h..16+8h+7}.
__device__ __forceinline__ v16h load_b_frag(const _Float16* tileT, int lane) {
  const int n = lane & 15, half = lane >> 4;
  const _Float16* p = tileT + n * LDK + half * 8;
  v8h lo = *(const v8h*)(p);
  v8h hi = *(const v8h*)(p + 16);
  v16h r;
#pragma unroll
  for (int j = 0; j < 8; ++j) { r[j] = lo[j]; r[j + 8] = hi[j]; }
  return r;
}

// ---------------- utility kernels ----------------

__global__ void k_zero_f4(float4* __restrict__ p, int n4) {
  int i = blockIdx.x * blockDim.x + threadIdx.x;
  if (i < n4) p[i] = make_float4(0.f, 0.f, 0.f, 0.f);
}

__global__ void k_cvt_f16(const float* __restrict__ s, _Float16* __restrict__ d, int n) {
  int i = (blockIdx.x * blockDim.x + threadIdx.x) * 4;
  if (i < n) {
    float4 v = *(const float4*)(s + i);
    v4h h; h[0] = (_Float16)v.x; h[1] = (_Float16)v.y; h[2] = (_Float16)v.z; h[3] = (_Float16)v.w;
    *(v4h*)(d + i) = h;
  }
}

// Transpose + convert: per expert z, src [R][C] f32 -> dst [C][R] f16.
__global__ __launch_bounds__(256)
void k_cvt_t(const float* __restrict__ s, _Float16* __restrict__ d, int R, int C) {
  __shared__ float tile[32][33];
  const int cb = blockIdx.x * 32, rb = blockIdx.y * 32;
  const float* sp = s + (size_t)blockIdx.z * R * C;
  _Float16* dp = d + (size_t)blockIdx.z * R * C;
  const int tx = threadIdx.x & 31, ty = threadIdx.x >> 5;  // 32 x 8
#pragma unroll
  for (int i = 0; i < 32; i += 8)
    tile[ty + i][tx] = sp[(size_t)(rb + ty + i) * C + cb + tx];
  __syncthreads();
#pragma unroll
  for (int i = 0; i < 32; i += 8)
    dp[(size_t)(cb + ty + i) * R + rb + tx] = (_Float16)tile[tx][ty + i];
}

// Deterministic routing: one block, 256 threads, 32 tokens per thread.
__global__ void k_route(const int* __restrict__ eidx, const float* __restrict__ ew,
                        int* __restrict__ counts, int* __restrict__ offsets,
                        int* __restrict__ slot_tok, float* __restrict__ slot_w) {
  __shared__ int cnt[256][N_EXPERTS];
  __shared__ int base[256][N_EXPERTS];
  __shared__ int eoff[N_EXPERTS];
  const int tid = threadIdx.x;
  const int t0 = tid * (N_TOKENS / 256);

  int c[N_EXPERTS];
#pragma unroll
  for (int e = 0; e < N_EXPERTS; ++e) c[e] = 0;
  for (int i = 0; i < N_TOKENS / 256; ++i) {
    int t = t0 + i;
    c[eidx[t * 2 + 0]]++;
    c[eidx[t * 2 + 1]]++;
  }
#pragma unroll
  for (int e = 0; e < N_EXPERTS; ++e) cnt[tid][e] = c[e];
  __syncthreads();

  if (tid == 0) {
    int acc[N_EXPERTS];
#pragma unroll
    for (int e = 0; e < N_EXPERTS; ++e) acc[e] = 0;
    for (int ch = 0; ch < 256; ++ch)
      for (int e = 0; e < N_EXPERTS; ++e) { base[ch][e] = acc[e]; acc[e] += cnt[ch][e]; }
    int o = 0;
    for (int e = 0; e < N_EXPERTS; ++e) {
      eoff[e] = o; offsets[e] = o; counts[e] = acc[e]; o += acc[e];
    }
  }
  __syncthreads();

  int cur[N_EXPERTS];
#pragma unroll
  for (int e = 0; e < N_EXPERTS; ++e) cur[e] = eoff[e] + base[tid][e];
  for (int i = 0; i < N_TOKENS / 256; ++i) {
    int t = t0 + i;
#pragma unroll
    for (int k = 0; k < TOP_K; ++k) {
      int e = eidx[t * 2 + k];
      int p = cur[e]++;
      slot_tok[p] = t;
      slot_w[p]   = ew[t * 2 + k];
    }
  }
}

// ---------------- FFN stage 1: h = silu(x@w1) * (x@w2), per expert ----------------
// grid: (SLOT_CAP/128, D_FF/64, E), block 256 (8 waves: 4(M) x 2(N))
// Double-buffered LDS: buffer = [As 128 | B1 64 | B2 64] rows x LDK halves.
#define FFN1_BUF (256 * LDK)
__global__ __launch_bounds__(256)
void k_ffn1(const _Float16* __restrict__ xh,
            const _Float16* __restrict__ w1t,  // [E][F][D] f16 (pre-transposed)
            const _Float16* __restrict__ w2t,  // [E][F][D] f16
            const int* __restrict__ counts, const int* __restrict__ offsets,
            const int* __restrict__ slot_tok,
            _Float16* __restrict__ hbuf) {
  const int e = blockIdx.z;
  const int cnt = counts[e];
  const int mBase = blockIdx.x * 128;
  if (mBase >= cnt) return;
  const int f0 = blockIdx.y * 64;
  const int slotBase = offsets[e];

  __shared__ _Float16 smem[2 * FFN1_BUF];  // 49152 B

  const int tid = threadIdx.x;
  const int lane = tid & 31;
  const int wid = tid >> 5;
  const int waveM = wid >> 1;   // 0..3 -> 32 rows
  const int waveN = wid & 1;    // 0..1 -> 32 cols

  // Loop-invariant staging: 4 x 16B per thread per K-chunk, uniform step along D.
  const _Float16* gsrc[4];
  int lofs[4];
#pragma unroll
  for (int it = 0; it < 2; ++it) {            // tasks 0..511: A tile (gathered rows)
    int task = tid + 256 * it;
    int row = task >> 2, q = task & 3;
    int r = mBase + row; if (r >= cnt) r = cnt - 1;   // clamp; rows >= cnt never stored
    int tok = slot_tok[slotBase + r];
    gsrc[it] = xh + (size_t)tok * D_MODEL + q * 8;
    lofs[it] = row * LDK + q * 8;
  }
  {                                           // tasks: B1 (64x4) and B2 (64x4)
    int fl = tid >> 2, q = tid & 3;
    size_t wb = (size_t)e * D_MODEL * D_FF + (size_t)(f0 + fl) * D_MODEL + q * 8;
    gsrc[2] = w1t + wb; lofs[2] = 128 * LDK + fl * LDK + q * 8;
    gsrc[3] = w2t + wb; lofs[3] = 192 * LDK + fl * LDK + q * 8;
  }

  const v8f vzero = {};
  v8f accG[2][2], accV[2][2];
#pragma unroll
  for (int mi = 0; mi < 2; ++mi)
#pragma unroll
    for (int ni = 0; ni < 2; ++ni) { accG[mi][ni] = vzero; accV[mi][ni] = vzero; }

  constexpr int NC = D_MODEL / 32;
  // Prologue: stage chunk 0 into buffer 0.
#pragma unroll
  for (int i = 0; i < 4; ++i) stage16(gsrc[i], &smem[lofs[i]]);

  for (int c = 0; c < NC; ++c) {
    const int boff = (c & 1) * FFN1_BUF;
    if (c + 1 < NC) {
      const int noff = ((c + 1) & 1) * FFN1_BUF;
      const int step = (c + 1) * 32;
#pragma unroll
      for (int i = 0; i < 4; ++i) stage16(gsrc[i] + step, &smem[noff + lofs[i]]);
      stage_wait_le4();    // chunk c landed; chunk c+1 still in flight
    } else {
      stage_wait_0();
    }
    __syncthreads();

    const _Float16* Asb = smem + boff;
    const _Float16* B1b = smem + boff + 128 * LDK;
    const _Float16* B2b = smem + boff + 192 * LDK;
    v16h afr[2], b1fr[2], b2fr[2];
#pragma unroll
    for (int mi = 0; mi < 2; ++mi) afr[mi] = load_a_frag(Asb + (waveM * 32 + mi * 16) * LDK, lane);
#pragma unroll
    for (int ni = 0; ni < 2; ++ni) {
      b1fr[ni] = load_b_frag(B1b + (waveN * 32 + ni * 16) * LDK, lane);
      b2fr[ni] = load_b_frag(B2b + (waveN * 32 + ni * 16) * LDK, lane);
    }
#pragma unroll
    for (int mi = 0; mi < 2; ++mi)
#pragma unroll
      for (int ni = 0; ni < 2; ++ni) {
        accG[mi][ni] = wmma_f16(afr[mi], b1fr[ni], accG[mi][ni]);
        accV[mi][ni] = wmma_f16(afr[mi], b2fr[ni], accV[mi][ni]);
      }
    __syncthreads();   // all waves done reading buf[c&1] before it is restaged
  }

  // Epilogue: h = silu(gate) * val -> f16. C/D layout: M = r + 8*(lane/16), N = lane%16.
  // sigmoid via single v_rcp_f32 instead of IEEE divide.
  const int half = lane >> 4, nl = lane & 15;
#pragma unroll
  for (int mi = 0; mi < 2; ++mi)
#pragma unroll
    for (int ni = 0; ni < 2; ++ni)
#pragma unroll
      for (int r = 0; r < 8; ++r) {
        int m = mBase + waveM * 32 + mi * 16 + half * 8 + r;
        if (m < cnt) {
          float g = accG[mi][ni][r];
          float v = accV[mi][ni][r];
          float sig = __builtin_amdgcn_rcpf(1.f + __expf(-g));
          float h = g * sig * v;
          hbuf[(size_t)(slotBase + m) * D_FF + f0 + waveN * 32 + ni * 16 + nl] = (_Float16)h;
        }
      }
}

// ---------------- FFN stage 2: out += cw * (h @ w3), per expert ----------------
// grid: (SLOT_CAP/128, D_MODEL/128, E), block 256 (8 waves: 4(M) x 2(N); wave 32x64)
// Double-buffered LDS: buffer = [As 128 | Bs 128] rows x LDK halves.
#define FFN2_BUF (256 * LDK)
__global__ __launch_bounds__(256)
void k_ffn2(const _Float16* __restrict__ hbuf,
            const _Float16* __restrict__ w3t,  // [E][D][F] f16 (pre-transposed)
            const int* __restrict__ counts, const int* __restrict__ offsets,
            const int* __restrict__ slot_tok, const float* __restrict__ slot_w,
            float* __restrict__ out) {
  const int e = blockIdx.z;
  const int cnt = counts[e];
  const int mBase = blockIdx.x * 128;
  if (mBase >= cnt) return;
  const int n0 = blockIdx.y * 128;
  const int slotBase = offsets[e];

  __shared__ _Float16 smem[2 * FFN2_BUF];  // 49152 B

  const int tid = threadIdx.x;
  const int lane = tid & 31;
  const int wid = tid >> 5;
  const int waveM = wid >> 1;
  const int waveN = wid & 1;

  const _Float16* gsrc[4];
  int lofs[4];
#pragma unroll
  for (int it = 0; it < 2; ++it) {
    int task = tid + 256 * it;
    int row = task >> 2, q = task & 3;
    int r = mBase + row; if (r >= cnt) r = cnt - 1;
    gsrc[it] = hbuf + (size_t)(slotBase + r) * D_FF + q * 8;
    lofs[it] = row * LDK + q * 8;
    int n = task >> 2;       // 0..127
    gsrc[2 + it] = w3t + (size_t)e * D_FF * D_MODEL + (size_t)(n0 + n) * D_FF + q * 8;
    lofs[2 + it] = 128 * LDK + n * LDK + q * 8;
  }

  const v8f vzero = {};
  v8f acc[2][4];
#pragma unroll
  for (int mi = 0; mi < 2; ++mi)
#pragma unroll
    for (int ni = 0; ni < 4; ++ni) acc[mi][ni] = vzero;

  constexpr int NC = D_FF / 32;
#pragma unroll
  for (int i = 0; i < 4; ++i) stage16(gsrc[i], &smem[lofs[i]]);

  for (int c = 0; c < NC; ++c) {
    const int boff = (c & 1) * FFN2_BUF;
    if (c + 1 < NC) {
      const int noff = ((c + 1) & 1) * FFN2_BUF;
      const int step = (c + 1) * 32;
#pragma unroll
      for (int i = 0; i < 4; ++i) stage16(gsrc[i] + step, &smem[noff + lofs[i]]);
      stage_wait_le4();
    } else {
      stage_wait_0();
    }
    __syncthreads();

    const _Float16* Asb = smem + boff;
    const _Float16* Bsb = smem + boff + 128 * LDK;
    v16h a[2], b[4];
#pragma unroll
    for (int mi = 0; mi < 2; ++mi) a[mi] = load_a_frag(Asb + (waveM * 32 + mi * 16) * LDK, lane);
#pragma unroll
    for (int ni = 0; ni < 4; ++ni) b[ni] = load_b_frag(Bsb + (waveN * 64 + ni * 16) * LDK, lane);
#pragma unroll
    for (int mi = 0; mi < 2; ++mi)
#pragma unroll
      for (int ni = 0; ni < 4; ++ni) acc[mi][ni] = wmma_f16(a[mi], b[ni], acc[mi][ni]);
    __syncthreads();
  }

  // Epilogue: scale by routing weight, accumulate into out (two slots per token).
  const int half = lane >> 4, nl = lane & 15;
#pragma unroll
  for (int mi = 0; mi < 2; ++mi)
#pragma unroll
    for (int r = 0; r < 8; ++r) {
      int m = mBase + waveM * 32 + mi * 16 + half * 8 + r;
      if (m < cnt) {
        int slot = slotBase + m;
        int tok = slot_tok[slot];
        float cw = slot_w[slot];
#pragma unroll
        for (int ni = 0; ni < 4; ++ni) {
          int n = n0 + waveN * 64 + ni * 16 + nl;
          atomicAdd(&out[(size_t)tok * D_MODEL + n], cw * acc[mi][ni][r]);
        }
      }
    }
}

// ---------------- launcher ----------------

extern "C" void kernel_launch(void* const* d_in, const int* in_sizes, int n_in,
                              void* d_out, int out_size, void* d_ws, size_t ws_size,
                              hipStream_t stream) {
  const float* x   = (const float*)d_in[0];
  const int*   idx = (const int*)d_in[1];
  const float* ew  = (const float*)d_in[2];
  const float* w1  = (const float*)d_in[3];
  const float* w2  = (const float*)d_in[4];
  const float* w3  = (const float*)d_in[5];
  float* out = (float*)d_out;

  char* ws = (char*)d_ws;
  size_t off = 0;
  auto take = [&](size_t bytes) -> char* {
    char* p = ws + off;
    off = (off + bytes + 255) & ~(size_t)255;
    return p;
  };

  const size_t nX = (size_t)N_TOKENS * D_MODEL;
  const size_t nW = (size_t)N_EXPERTS * D_MODEL * D_FF;
  _Float16* xh   = (_Float16*)take(nX * 2);
  _Float16* w1t  = (_Float16*)take(nW * 2);
  _Float16* w2t  = (_Float16*)take(nW * 2);
  _Float16* w3t  = (_Float16*)take(nW * 2);
  _Float16* hbuf = (_Float16*)take((size_t)(SLOT_CAP + 128) * D_FF * 2);
  int*   counts   = (int*)take(N_EXPERTS * sizeof(int));
  int*   offsets  = (int*)take(N_EXPERTS * sizeof(int));
  int*   slot_tok = (int*)take(SLOT_CAP * sizeof(int));
  float* slot_w   = (float*)take(SLOT_CAP * sizeof(float));
  (void)ws_size; (void)n_in; (void)in_sizes; (void)out_size;

  // 1) zero output (we accumulate atomically)
  k_zero_f4<<<(int)(nX / 4 / 256), 256, 0, stream>>>((float4*)out, (int)(nX / 4));

  // 2) conversions: x row-major f16; weights transposed f16 (L2-resident afterwards)
  k_cvt_f16<<<(int)(nX / 4 / 256), 256, 0, stream>>>(x, xh, (int)nX);
  {
    dim3 gw12(D_FF / 32, D_MODEL / 32, N_EXPERTS);   // src [D][F] -> dst [F][D]
    k_cvt_t<<<gw12, 256, 0, stream>>>(w1, w1t, D_MODEL, D_FF);
    k_cvt_t<<<gw12, 256, 0, stream>>>(w2, w2t, D_MODEL, D_FF);
    dim3 gw3(D_MODEL / 32, D_FF / 32, N_EXPERTS);    // src [F][D] -> dst [D][F]
    k_cvt_t<<<gw3, 256, 0, stream>>>(w3, w3t, D_FF, D_MODEL);
  }

  // 3) deterministic routing
  k_route<<<1, 256, 0, stream>>>(idx, ew, counts, offsets, slot_tok, slot_w);

  // 4) expert FFN stage 1 (worst-case grid; early-exit on counts[e])
  dim3 g1(SLOT_CAP / 128, D_FF / 64, N_EXPERTS);
  k_ffn1<<<g1, 256, 0, stream>>>(xh, w1t, w2t, counts, offsets, slot_tok, hbuf);

  // 5) expert FFN stage 2 + weighted scatter-add
  dim3 g2(SLOT_CAP / 128, D_MODEL / 128, N_EXPERTS);
  k_ffn2<<<g2, 256, 0, stream>>>(hbuf, w3t, counts, offsets, slot_tok, slot_w, out);
}
